// PairMap_25786983645305
// MI455X (gfx1250) — matrix operand order repaired and checked by reading, not compiled, not run on hardware
//
#include <hip/hip_runtime.h>
#include <stdint.h>

// ---------------- problem constants ----------------
#define BATCH   32
#define NA      128
#define NBND    1024
#define DF      128     // per-atom feature
#define DF2     256     // concat feature
#define HID     512
#define DOUT    128
#define NT      5
#define NPAIR   25
#define NBOND   (BATCH * NBND)   // 32768

// M tile per workgroup (2 row-tiles of 16, 8 waves)
#define MB      32
#define XPITCH  258              // bf16 elems, bank-conflict-free (516B row stride)
#define HPITCH  514              // bf16 elems (1028B row stride)

// fragment-major weight tensors
#define KT1     (DF2 / 32)       // 8  k-tiles, layer 1
#define NT1     (HID / 16)       // 32 n-tiles, layer 1
#define KT2     (HID / 32)       // 16 k-tiles, layer 2
#define NT2     (DOUT / 16)      // 8  n-tiles, layer 2
#define FRAGU32 256              // 32 lanes x 8 dwords per 32x16 bf16 B fragment

typedef __attribute__((ext_vector_type(16))) __bf16 bf16x16;
typedef __attribute__((ext_vector_type(8)))  float  floatx8;

union FragB16 { bf16x16 v; uint32_t u[8]; };

// round-to-nearest-even f32 -> bf16 bit pattern
__device__ __forceinline__ uint16_t f2bf(float x) {
    uint32_t u = __float_as_uint(x);
    u += 0x7FFFu + ((u >> 16) & 1u);
    return (uint16_t)(u >> 16);
}
__device__ __forceinline__ uint32_t pack2bf(float lo, float hi) {
    return (uint32_t)f2bf(lo) | ((uint32_t)f2bf(hi) << 16);
}

// ---------------------------------------------------------------------------
// Kernel 1: convert W1/W2 to bf16 and store FRAGMENT-MAJOR: for each
// (type, ktile, ntile), the exact 32-lane x 8-dword WMMA B operand image,
// each lane's 8 dwords contiguous -> B fragment = two b128 loads per lane.
// B 32x16 bf16 layout: lanes 0-15 (N=lane) hold K {0..7,16..23}, lanes 16-31
// hold K {8..15,24..31}; K-pairs per VGPR.  Also zero bucket counters.
// ---------------------------------------------------------------------------
__global__ void pack_weights(const float* __restrict__ W1,
                             const float* __restrict__ W2,
                             uint32_t* __restrict__ w1f,
                             uint32_t* __restrict__ w2f,
                             int* __restrict__ counts) {
    int tid = blockIdx.x * blockDim.x + threadIdx.x;
    if (tid < 32) counts[tid] = 0;

    const int n1 = NPAIR * KT1 * NT1 * FRAGU32;   // 1,638,400
    const int n2 = NPAIR * KT2 * NT2 * FRAGU32;   //   819,200
    for (int e = tid; e < n1 + n2; e += gridDim.x * blockDim.x) {
        if (e < n1) {
            // per-type span = 8*32*256 = 65536
            int v    = e & 7;
            int lane = (e >> 3) & 31;
            int nt   = (e >> 8) & 31;
            int kt   = (e >> 13) & 7;
            int p    = e >> 16;
            int k    = kt * 32 + ((v >> 2) << 4) + ((lane & 16) ? 8 : 0) + ((v & 3) << 1);
            int n    = nt * 16 + (lane & 15);
            const float* base = W1 + (size_t)p * DF2 * HID;
            w1f[e] = pack2bf(base[k * HID + n], base[(k + 1) * HID + n]);
        } else {
            int e2   = e - n1;
            // per-type span = 16*8*256 = 32768
            int v    = e2 & 7;
            int lane = (e2 >> 3) & 31;
            int nt   = (e2 >> 8) & 7;
            int kt   = (e2 >> 11) & 15;
            int p    = e2 >> 15;
            int k    = kt * 32 + ((v >> 2) << 4) + ((lane & 16) ? 8 : 0) + ((v & 3) << 1);
            int n    = nt * 16 + (lane & 15);
            const float* base = W2 + (size_t)p * HID * DOUT;
            w2f[e2] = pack2bf(base[k * DOUT + n], base[(k + 1) * DOUT + n]);
        }
    }
}

// ---------------------------------------------------------------------------
// Kernel 2: per-bond prep — normalized bond vectors, zero y for padded rows,
// bucket valid rows by pair type (atomic counters + row lists in workspace).
// ---------------------------------------------------------------------------
__global__ void prep(const int* __restrict__ z, const float* __restrict__ r,
                     const int* __restrict__ t,
                     int* __restrict__ counts, int* __restrict__ rowlist,
                     float* __restrict__ y_out, float* __restrict__ v_out) {
    int n = blockIdx.x * blockDim.x + threadIdx.x;
    if (n >= NBOND) return;
    int t1 = t[2 * n], t2 = t[2 * n + 1];
    int b  = n / NBND;
    bool valid = (t1 != -1);
    int g1 = b * NA + (t1 < 0 ? 0 : t1);
    int g2 = b * NA + (t2 < 0 ? 0 : t2);
    float vx = 0.f, vy = 0.f, vz = 0.f;
    if (valid) {
        float dx = r[g2 * 3 + 0] - r[g1 * 3 + 0];
        float dy = r[g2 * 3 + 1] - r[g1 * 3 + 1];
        float dz = r[g2 * 3 + 2] - r[g1 * 3 + 2];
        float n2  = dx * dx + dy * dy + dz * dz;
        float inv = rsqrtf(fmaxf(n2, 1e-24f));
        vx = dx * inv; vy = dy * inv; vz = dz * inv;
        int idx = z[g1] * NT + z[g2];
        int pos = atomicAdd(&counts[idx], 1);
        rowlist[idx * NBOND + pos] = n;
    } else {
        float4 zero{0.f, 0.f, 0.f, 0.f};
        float4* yp = (float4*)(y_out + (size_t)n * DOUT);
#pragma unroll
        for (int i = 0; i < DOUT / 4; ++i) yp[i] = zero;
    }
    v_out[3 * n + 0] = vx;
    v_out[3 * n + 1] = vy;
    v_out[3 * n + 2] = vz;
}

// ---------------------------------------------------------------------------
// Fragment loaders
// A 16x32 bf16 (from LDS): lanes 0-15 M=0..15 hold K {0..7,16..23};
// lanes 16-31 same M, K {8..15,24..31}; K-pairs per VGPR.
// B (from global, fragment-major): lane's 8 dwords contiguous -> 2x b128.
// ---------------------------------------------------------------------------
__device__ __forceinline__ bf16x16 load_a_lds(const unsigned short* rowp,
                                              int K0, int khA) {
    FragB16 f;
    const uint32_t* p = (const uint32_t*)rowp;  // row base is 4B aligned
#pragma unroll
    for (int v = 0; v < 8; ++v) {
        int k = K0 + ((v >> 2) << 4) + khA + ((v & 3) << 1);
        f.u[v] = p[k >> 1];
    }
    return f.v;
}
__device__ __forceinline__ bf16x16 load_b_frag(const uint32_t* __restrict__ fp) {
    FragB16 f;
    const uint4* p = (const uint4*)fp;          // 32B-aligned (index % 8 == 0)
    uint4 a = p[0], b = p[1];
    f.u[0] = a.x; f.u[1] = a.y; f.u[2] = a.z; f.u[3] = a.w;
    f.u[4] = b.x; f.u[5] = b.y; f.u[6] = b.z; f.u[7] = b.w;
    return f.v;
}

// ---------------------------------------------------------------------------
// Kernel 3: grouped pair-type MLP via WMMA bf16 (fp32 accumulate).
// grid = NPAIR * (NBOND/MB); blocks early-exit past their bucket's count.
// 256 threads = 8 waves; 2 row-tiles (16 rows) x 4 column groups.
// ---------------------------------------------------------------------------
__global__ __launch_bounds__(256) void pair_mlp(
        const int* __restrict__ t,
        const int* __restrict__ counts, const int* __restrict__ rowlist,
        const uint32_t* __restrict__ w1f, const uint32_t* __restrict__ w2f,
        const float* __restrict__ b1, const float* __restrict__ b2,
        const float* __restrict__ xg, float* __restrict__ y_out) {
    const int type = blockIdx.x / (NBOND / MB);
    const int mblk = blockIdx.x % (NBOND / MB);
    const int cnt  = counts[type];
    const int base = mblk * MB;
    if (base >= cnt) return;
    const int vslots = (cnt - base < MB) ? (cnt - base) : MB;

    __shared__ int g1s[MB], g2s[MB], rgs[MB];
    __shared__ __align__(8) unsigned short xcs[MB * XPITCH];  // x_c tile, bf16
    __shared__ __align__(8) unsigned short hs[MB * HPITCH];   // hidden, bf16

    const int tid = threadIdx.x;
    if (tid < MB) {
        int rrow = base + tid;
        int rg   = rowlist[type * NBOND + (rrow < cnt ? rrow : base)];
        rgs[tid] = rg;
        int b    = rg / NBND;
        g1s[tid] = b * NA + t[2 * rg];
        g2s[tid] = b * NA + t[2 * rg + 1];
    }
    __syncthreads();

    // gather concat(x[t1], x[t2]) -> LDS bf16 (coalesced along feature dim)
    for (int i = tid; i < MB * DF2; i += 256) {
        int row = i >> 8, col = i & 255;
        int g   = (col < DF) ? g1s[row] : g2s[row];
        float xv = xg[(size_t)g * DF + (col & (DF - 1))];
        xcs[row * XPITCH + col] = f2bf(xv);
    }
    __syncthreads();

    const int w    = tid >> 5;
    const int lane = tid & 31;
    const int rt   = w >> 2;            // row-tile 0..1
    const int q    = w & 3;             // column quarter 0..3
    const int khA  = (lane & 16) ? 8 : 0;
    const int nc   = lane & 15;         // N within tile (B/C/D layouts)
    const int arow = rt * 16 + nc;      // M within tile (A layout)
    const int mb2  = (lane >> 4) << 3;  // C/D: lanes 16-31 hold M+8

    // hoist the 8 K-tile A-fragments for this wave's row-tile
    bf16x16 afrag[8];
#pragma unroll
    for (int kt = 0; kt < 8; ++kt)
        afrag[kt] = load_a_lds(&xcs[arow * XPITCH], kt * 32, khA);

    // -------- layer 1: H[32x512] = silu(Xc @ W1 + b1), this wave: 8 n-tiles
    // fragment-major base for this type + this lane
    const uint32_t* w1base = w1f + (size_t)type * (KT1 * NT1 * FRAGU32) + lane * 8;
#pragma unroll
    for (int c = 0; c < 2; ++c) {
        floatx8 acc[4];
#pragma unroll
        for (int j = 0; j < 4; ++j)
#pragma unroll
            for (int e = 0; e < 8; ++e) acc[j][e] = 0.f;

#pragma unroll
        for (int kt = 0; kt < 8; ++kt) {
#pragma unroll
            for (int j = 0; j < 4; ++j) {
                int ntg = q * 8 + c * 4 + j;
                bf16x16 bfr = load_b_frag(w1base + (kt * NT1 + ntg) * FRAGU32);
                acc[j] = __builtin_amdgcn_wmma_f32_16x16x32_bf16(
                    false, afrag[kt], false, bfr, (short)0, acc[j], false, false);
            }
        }
#pragma unroll
        for (int j = 0; j < 4; ++j) {
            int col   = (q * 8 + c * 4 + j) * 16 + nc;
            float bia = b1[type * HID + col];
#pragma unroll
            for (int rI = 0; rI < 8; ++rI) {
                float v = acc[j][rI] + bia;
                float s = v / (1.f + __expf(-v));   // silu
                hs[(rt * 16 + mb2 + rI) * HPITCH + col] = f2bf(s);
            }
        }
    }
    __syncthreads();

    // -------- layer 2: Y[32x128] = H @ W2 + b2, this wave: 2 n-tiles
    const uint32_t* w2base = w2f + (size_t)type * (KT2 * NT2 * FRAGU32) + lane * 8;
    floatx8 acc2[2];
#pragma unroll
    for (int j = 0; j < 2; ++j)
#pragma unroll
        for (int e = 0; e < 8; ++e) acc2[j][e] = 0.f;

#pragma unroll
    for (int kt = 0; kt < 16; ++kt) {
        bf16x16 a2 = load_a_lds(&hs[arow * HPITCH], kt * 32, khA);
#pragma unroll
        for (int j = 0; j < 2; ++j) {
            int ntg = q * 2 + j;
            bf16x16 bfr = load_b_frag(w2base + (kt * NT2 + ntg) * FRAGU32);
            acc2[j] = __builtin_amdgcn_wmma_f32_16x16x32_bf16(
                false, a2, false, bfr, (short)0, acc2[j], false, false);
        }
    }
#pragma unroll
    for (int j = 0; j < 2; ++j) {
        int col   = (q * 2 + j) * 16 + nc;
        float bia = b2[type * DOUT + col];
#pragma unroll
        for (int rI = 0; rI < 8; ++rI) {
            int slot = rt * 16 + mb2 + rI;
            if (slot < vslots)
                y_out[(size_t)rgs[slot] * DOUT + col] = acc2[j][rI] + bia;
        }
    }
}

// ---------------------------------------------------------------------------
// Host-side launch
// ---------------------------------------------------------------------------
extern "C" void kernel_launch(void* const* d_in, const int* in_sizes, int n_in,
                              void* d_out, int out_size, void* d_ws, size_t ws_size,
                              hipStream_t stream) {
    (void)in_sizes; (void)n_in; (void)out_size; (void)ws_size;
    const int*   z  = (const int*)d_in[0];
    const float* r  = (const float*)d_in[1];
    const float* x  = (const float*)d_in[2];
    const int*   t  = (const int*)d_in[3];
    const float* W1 = (const float*)d_in[4];
    const float* b1 = (const float*)d_in[5];
    const float* W2 = (const float*)d_in[6];
    const float* b2 = (const float*)d_in[7];

    float* out = (float*)d_out;
    // output layout (flat element offsets): z | x | t | y | v
    const size_t OFF_Z = 0;
    const size_t OFF_X = OFF_Z + (size_t)BATCH * NA;                 // 4096
    const size_t OFF_T = OFF_X + (size_t)BATCH * NA * DF;            // 528384
    const size_t OFF_Y = OFF_T + (size_t)BATCH * NBND * 2;           // 593920
    const size_t OFF_V = OFF_Y + (size_t)BATCH * NBND * DOUT;        // 4788224

    // workspace layout (16B-aligned sections)
    char* ws = (char*)d_ws;
    int*      counts  = (int*)ws;                                    // 32 ints
    int*      rowlist = (int*)(ws + 128);                            // 25*32768 ints
    uint32_t* w1f     = (uint32_t*)(ws + 128 + (size_t)NPAIR * NBOND * 4);
    uint32_t* w2f     = w1f + (size_t)NPAIR * KT1 * NT1 * FRAGU32;

    // passthrough copies (graph-capture safe async d2d)
    hipMemcpyAsync(out + OFF_Z, z, (size_t)BATCH * NA * 4,        hipMemcpyDeviceToDevice, stream);
    hipMemcpyAsync(out + OFF_X, x, (size_t)BATCH * NA * DF * 4,   hipMemcpyDeviceToDevice, stream);
    hipMemcpyAsync(out + OFF_T, t, (size_t)BATCH * NBND * 2 * 4,  hipMemcpyDeviceToDevice, stream);

    pack_weights<<<4096, 256, 0, stream>>>(W1, W2, w1f, w2f, counts);
    prep<<<NBOND / 256, 256, 0, stream>>>(z, r, t, counts, rowlist,
                                          out + OFF_Y, out + OFF_V);
    pair_mlp<<<NPAIR * (NBOND / MB), 256, 0, stream>>>(
        t, counts, rowlist, w1f, w2f, b1, b2, x, out + OFF_Y);
}
